// LorentzParT_31851477467192
// MI455X (gfx1250) — matrix operand order repaired and checked by reading, not compiled, not run on hardware
//
#include <hip/hip_runtime.h>
#include <hip/hip_bf16.h>
#include <cstdint>
#include <cstddef>

typedef _Float16 half_t;
typedef _Float16 v16h __attribute__((ext_vector_type(16)));
typedef _Float16 v8h  __attribute__((ext_vector_type(8)));
typedef float    v8f  __attribute__((ext_vector_type(8)));
typedef unsigned int u32x4 __attribute__((ext_vector_type(4)));
typedef int          i32x4 __attribute__((ext_vector_type(4)));
typedef int          i32x8 __attribute__((ext_vector_type(8)));

#define BB     64
#define LLEN   128
#define DD     512
#define HHN    8
#define NLAYER 8
#define FEATN  4
#define NCLSN  10
#define HDD    64
#define D3     1536
#define D4     2048
#define DHALF  256
#define DQRT   128

// ---------------------------------------------------------------------------
// device helpers
// ---------------------------------------------------------------------------
__device__ __forceinline__ float gelu_exact(float x) {
    return 0.5f * x * (1.0f + erff(x * 0.70710678118654752f));
}
__device__ __forceinline__ float softplus_f(float x) {
    return (x > 20.0f) ? x : log1pf(expf(x));
}

// A-matrix fragment (16x32 f16, MxK).  lane<16: K 0..7 & 16..23 ; lane>=16: K 8..15 & 24..31
__device__ __forceinline__ v16h load_a_frag(const half_t* A, int lda, int row0, int k0) {
    int lane = threadIdx.x & 31;
    const half_t* p = A + (size_t)(row0 + (lane & 15)) * lda + k0 + ((lane >> 4) << 3);
    v8h lo = *(const v8h*)(p);
    v8h hi = *(const v8h*)(p + 16);
    v16h r;
#pragma unroll
    for (int i = 0; i < 8; ++i) { r[i] = lo[i]; r[i + 8] = hi[i]; }
    return r;
}

// B-matrix fragment (32x16 f16, KxN) from W stored as N x K row-major.
// lane<16: K 0..7 & 16..23 of column (lane), lane>=16: K 8..15 & 24..31
__device__ __forceinline__ v16h load_b_frag(const half_t* W, int ldw, int n0, int k0) {
    int lane = threadIdx.x & 31;
    const half_t* p = W + (size_t)(n0 + (lane & 15)) * ldw + k0 + ((lane >> 4) << 3);
    v8h lo = *(const v8h*)(p);
    v8h hi = *(const v8h*)(p + 16);
    v16h r;
#pragma unroll
    for (int i = 0; i < 8; ++i) { r[i] = lo[i]; r[i + 8] = hi[i]; }
    return r;
}

__device__ __forceinline__ v8f wmma_f16(v16h a, v16h b, v8f c) {
    return __builtin_amdgcn_wmma_f32_16x16x32_f16(false, a, false, b, (short)0, c, false, false);
}

// ---------------------------------------------------------------------------
// Tensor Data Mover: async-DMA a 2D tile (tile_n rows x tile_k f16) of a
// row-major (tensor_n x tensor_k, stride ldw) f16 matrix into LDS (packed
// tile_n x tile_k).  Descriptor layout per CDNA5 ISA section 8 (D# groups).
// ---------------------------------------------------------------------------
__device__ __forceinline__ void tdm_load_tile_f16(const half_t* gsrc, unsigned lds_byte_off,
                                                  unsigned tensor_k, unsigned tensor_n,
                                                  unsigned ldw, unsigned tile_k, unsigned tile_n) {
    unsigned long long ga = (unsigned long long)(uintptr_t)gsrc;
    u32x4 g0;
    g0[0] = 1u;                                            // count=1 valid descriptor
    g0[1] = lds_byte_off;                                  // lds_addr (bytes)
    g0[2] = (unsigned)(ga & 0xffffffffull);                // global_addr[31:0]
    g0[3] = (unsigned)((ga >> 32) & 0x1ffffffull) | (2u << 30); // addr[56:32] | type=2
    i32x8 g1;
    unsigned long long s0 = (unsigned long long)ldw;       // dim0 stride (elements)
    g1[0] = 0x00010000;                                    // wg_mask=0, data_size=1 (2 bytes)
    g1[1] = (int)((tensor_k & 0xffffu) << 16);             // atomic_addr=0 | tensor_dim0 lo16
    g1[2] = (int)((tensor_k >> 16) | ((tensor_n & 0xffffu) << 16)); // dim0 hi | dim1 lo
    g1[3] = (int)((tensor_n >> 16) | (tile_k << 16));      // dim1 hi | tile_dim0
    g1[4] = (int)(tile_n & 0xffffu);                       // tile_dim1 | tile_dim2=0
    g1[5] = (int)(s0 & 0xffffffffull);                     // dim0_stride lo32
    g1[6] = (int)((s0 >> 32) & 0xffffull);                 // dim0_stride hi16 | dim1_stride lo=0
    g1[7] = 0;
    i32x4 gz = {0, 0, 0, 0};
#if __clang_major__ >= 23
    i32x8 gz8 = {0, 0, 0, 0, 0, 0, 0, 0};
    __builtin_amdgcn_tensor_load_to_lds(g0, g1, gz, gz, gz8, 0);
#else
    __builtin_amdgcn_tensor_load_to_lds(g0, g1, gz, gz, 0);
#endif
}

// ---------------------------------------------------------------------------
// TDM-staged WMMA GEMM:  C[M,N] = act( A[M,K] * W[N,K]^T + bias + residual )
// block = 8 waves = 256x64 output tile; wave w owns rows [w*32, w*32+32).
// Weight tile (64 x 64 f16) is double-buffered in LDS by the Tensor Data
// Mover (wave 0 issues, TENSORcnt + barrier synchronize), WMMA B-fragments
// are read from LDS, A streams from global.
// Requires: M%256==0, N%64==0, K%64==0.
// ---------------------------------------------------------------------------
__global__ void k_gemm_tdm(const half_t* __restrict__ A, const half_t* __restrict__ W,
                           const float* __restrict__ bias, const float* __restrict__ res,
                           float* __restrict__ Cf, half_t* __restrict__ Ch,
                           int M, int N, int K, int lda, int ldw, int ldc, int act) {
    __shared__ half_t sB[2][64 * 64];   // 16 KB double buffer
    int bN = N >> 6;
    int bm = blockIdx.x / bN;
    int bn = blockIdx.x % bN;
    int wave = threadIdx.x >> 5;
    int lane = threadIdx.x & 31;
    int m0 = (bm << 8) + (wave << 5);
    int n0 = bn << 6;

    int nChunks = K >> 6;

    if (wave == 0) {
        tdm_load_tile_f16(W + (size_t)n0 * ldw, (unsigned)(uintptr_t)&sB[0][0],
                          (unsigned)K, (unsigned)N - n0, (unsigned)ldw, 64u, 64u);
    }

    v8f acc[2][4];
#pragma unroll
    for (int i = 0; i < 2; ++i)
#pragma unroll
        for (int j = 0; j < 4; ++j) acc[i][j] = (v8f){};

    for (int c = 0; c < nChunks; ++c) {
        int buf = c & 1;
        if (wave == 0) {
            __builtin_amdgcn_s_wait_tensorcnt(0);   // chunk c landed in LDS
        }
        __syncthreads();
        if (wave == 0 && (c + 1) < nChunks) {
            tdm_load_tile_f16(W + (size_t)n0 * ldw + ((c + 1) << 6),
                              (unsigned)(uintptr_t)&sB[buf ^ 1][0],
                              (unsigned)K - ((c + 1) << 6), (unsigned)N - n0,
                              (unsigned)ldw, 64u, 64u);
        }
        const half_t* sBp = &sB[buf][0];
        int kc = c << 6;
#pragma unroll
        for (int step = 0; step < 2; ++step) {
            int k0g = kc + step * 32;
            int k0l = step * 32;
            if (k0g + 32 < K) {
                __builtin_prefetch(A + (size_t)(m0 + (lane & 15)) * lda + k0g + 32, 0, 1);
            }
            v16h a0 = load_a_frag(A, lda, m0, k0g);
            v16h a1 = load_a_frag(A, lda, m0 + 16, k0g);
            v16h b0 = load_b_frag(sBp, 64, 0,  k0l);
            v16h b1 = load_b_frag(sBp, 64, 16, k0l);
            v16h b2 = load_b_frag(sBp, 64, 32, k0l);
            v16h b3 = load_b_frag(sBp, 64, 48, k0l);
            acc[0][0] = wmma_f16(a0, b0, acc[0][0]);
            acc[0][1] = wmma_f16(a0, b1, acc[0][1]);
            acc[0][2] = wmma_f16(a0, b2, acc[0][2]);
            acc[0][3] = wmma_f16(a0, b3, acc[0][3]);
            acc[1][0] = wmma_f16(a1, b0, acc[1][0]);
            acc[1][1] = wmma_f16(a1, b1, acc[1][1]);
            acc[1][2] = wmma_f16(a1, b2, acc[1][2]);
            acc[1][3] = wmma_f16(a1, b3, acc[1][3]);
        }
        __syncthreads();   // all waves done reading buf before it is re-filled
    }

#pragma unroll
    for (int ti = 0; ti < 2; ++ti)
#pragma unroll
        for (int tj = 0; tj < 4; ++tj) {
            int n = n0 + tj * 16 + (lane & 15);
            int mb = m0 + ti * 16 + ((lane >> 4) << 3);
            float bv = bias ? bias[n] : 0.0f;
#pragma unroll
            for (int v = 0; v < 8; ++v) {
                float val = acc[ti][tj][v] + bv;
                size_t idx = (size_t)(mb + v) * ldc + n;
                if (res) val += res[idx];
                if (act == 1) val = gelu_exact(val);
                if (Ch) Ch[idx] = (half_t)val;
                else    Cf[idx] = val;
            }
        }
}

// ---------------------------------------------------------------------------
// batched WMMA GEMM for attention (per (b,h) slice), C = scale * A * W^T
// ---------------------------------------------------------------------------
__global__ void k_gemm_batched(const half_t* __restrict__ A, const half_t* __restrict__ Bw,
                               float* __restrict__ Cf, half_t* __restrict__ Ch,
                               int M, int N, int K, int lda, int ldb, int ldc,
                               long sAb, long sAh, long sBb, long sBh, long sCb, long sCh,
                               int Hn, float scale) {
    int z = blockIdx.z;
    int b = z / Hn, hh = z % Hn;
    const half_t* Ap = A + (size_t)b * sAb + (size_t)hh * sAh;
    const half_t* Bp = Bw + (size_t)b * sBb + (size_t)hh * sBh;
    size_t coff = (size_t)b * sCb + (size_t)hh * sCh;

    int wave = blockIdx.x * (blockDim.x >> 5) + (threadIdx.x >> 5);
    int lane = threadIdx.x & 31;
    int tilesN = N >> 5;
    int tilesM = M >> 5;
    if (wave >= tilesM * tilesN) return;
    int tm = wave / tilesN, tn = wave % tilesN;
    int m0 = tm << 5, n0 = tn << 5;

    v8f acc[2][2];
#pragma unroll
    for (int i = 0; i < 2; ++i)
#pragma unroll
        for (int j = 0; j < 2; ++j) acc[i][j] = (v8f){};

    for (int k0 = 0; k0 < K; k0 += 32) {
        v16h a0 = load_a_frag(Ap, lda, m0, k0);
        v16h a1 = load_a_frag(Ap, lda, m0 + 16, k0);
        v16h b0 = load_b_frag(Bp, ldb, n0, k0);
        v16h b1 = load_b_frag(Bp, ldb, n0 + 16, k0);
        acc[0][0] = wmma_f16(a0, b0, acc[0][0]);
        acc[0][1] = wmma_f16(a0, b1, acc[0][1]);
        acc[1][0] = wmma_f16(a1, b0, acc[1][0]);
        acc[1][1] = wmma_f16(a1, b1, acc[1][1]);
    }

#pragma unroll
    for (int ti = 0; ti < 2; ++ti)
#pragma unroll
        for (int tj = 0; tj < 2; ++tj) {
            int n = n0 + tj * 16 + (lane & 15);
            int mb = m0 + ti * 16 + ((lane >> 4) << 3);
#pragma unroll
            for (int v = 0; v < 8; ++v) {
                float val = acc[ti][tj][v] * scale;
                size_t idx = coff + (size_t)(mb + v) * ldc + n;
                if (Ch) Ch[idx] = (half_t)val;
                else    Cf[idx] = val;
            }
        }
}

// ---------------------------------------------------------------------------
// elementwise / reduction kernels
// ---------------------------------------------------------------------------
__global__ void k_f32_to_f16(const float* __restrict__ src, half_t* __restrict__ dst, int n) {
    int i = blockIdx.x * blockDim.x + threadIdx.x;
    if (i < n) dst[i] = (half_t)src[i];
}

__global__ void k_interaction(const float* __restrict__ x, float* __restrict__ inter) {
    int tid = blockIdx.x * blockDim.x + threadIdx.x;
    if (tid >= BB * LLEN * LLEN) return;
    int j = tid % LLEN;
    int r = tid / LLEN;
    int i = r % LLEN;
    int b = r / LLEN;
    const float* xb = x + (size_t)b * FEATN * LLEN;
    float pTi = fmaxf(xb[0 * LLEN + i], 0.f), pTj = fmaxf(xb[0 * LLEN + j], 0.f);
    float etai = xb[1 * LLEN + i], etaj = xb[1 * LLEN + j];
    float phii = xb[2 * LLEN + i], phij = xb[2 * LLEN + j];
    float Ei = fmaxf(xb[3 * LLEN + i], 0.f), Ej = fmaxf(xb[3 * LLEN + j], 0.f);
    float deta = etai - etaj;
    float dp = phii - phij;
    float dphi = atan2f(sinf(dp), cosf(dp));
    float dR = sqrtf(deta * deta + dphi * dphi + 1e-8f);
    float mn = fminf(pTi, pTj);
    float kT = mn * dR;
    float zf = mn / (pTi + pTj + 1e-8f);
    float esi = fminf(fmaxf(etai, -5.f), 5.f);
    float esj = fminf(fmaxf(etaj, -5.f), 5.f);
    float pxi = pTi * cosf(phii), pyi = pTi * sinf(phii), pzi = pTi * sinhf(esi);
    float pxj = pTj * cosf(phij), pyj = pTj * sinf(phij), pzj = pTj * sinhf(esj);
    float se = Ei + Ej, sx = pxi + pxj, sy = pyi + pyj, sz = pzi + pzj;
    float m2 = fmaxf(se * se - sx * sx - sy * sy - sz * sz, 1e-6f);
    float o0 = logf(fmaxf(dR, 1e-8f));
    float o1 = logf(fmaxf(kT, 1e-8f));
    float o2 = logf(fmaxf(zf, 1e-8f));
    float o3 = logf(m2);
    if (o0 != o0) o0 = 0.f;
    if (o1 != o1) o1 = 0.f;
    if (o2 != o2) o2 = 0.f;
    if (o3 != o3) o3 = 0.f;
    float* o = inter + (size_t)tid * 4;
    o[0] = o0; o[1] = o1; o[2] = o2; o[3] = o3;
}

__global__ void k_embed(const float* __restrict__ x, const float* __restrict__ Win,
                        const float* __restrict__ bin, const float* __restrict__ pos,
                        float* __restrict__ h) {
    int tid = blockIdx.x * blockDim.x + threadIdx.x;
    if (tid >= BB * LLEN * DD) return;
    int d = tid % DD;
    int r = tid / DD;
    int l = r % LLEN;
    int b = r / LLEN;
    const float* xb = x + (size_t)b * FEATN * LLEN;
    float acc = bin[d] + pos[(size_t)l * DD + d];
#pragma unroll
    for (int f = 0; f < FEATN; ++f) acc += xb[f * LLEN + l] * Win[d * FEATN + f];
    h[tid] = acc;
}

__global__ void k_ln(const float* __restrict__ in, const float* __restrict__ g,
                     const float* __restrict__ bp, half_t* __restrict__ out) {
    int row = blockIdx.x;
    int t = threadIdx.x; // 256
    const float* xr = in + (size_t)row * DD;
    float x0 = xr[t], x1 = xr[t + 256];
    __shared__ float red[256];
    red[t] = x0 + x1;
    __syncthreads();
    for (int s = 128; s > 0; s >>= 1) { if (t < s) red[t] += red[t + s]; __syncthreads(); }
    float mean = red[0] * (1.0f / DD);
    __syncthreads();
    float d0 = x0 - mean, d1 = x1 - mean;
    red[t] = d0 * d0 + d1 * d1;
    __syncthreads();
    for (int s = 128; s > 0; s >>= 1) { if (t < s) red[t] += red[t + s]; __syncthreads(); }
    float rs = rsqrtf(red[0] * (1.0f / DD) + 1e-5f);
    out[(size_t)row * DD + t]       = (half_t)(d0 * rs * g[t] + bp[t]);
    out[(size_t)row * DD + t + 256] = (half_t)(d1 * rs * g[t + 256] + bp[t + 256]);
}

__global__ void k_equi_finish(const float* __restrict__ vec, const float* __restrict__ hin,
                              const float* __restrict__ Ws, const float* __restrict__ bs,
                              const float* __restrict__ g, const float* __restrict__ bp,
                              float* __restrict__ hout) {
    int row = blockIdx.x;
    int t = threadIdx.x; // 256
    __shared__ float s_sh;
    const float* hv = hin + (size_t)row * DD;
    if (t == 0) {
        float a = hv[0], b2 = hv[1], c = hv[2], d = hv[3];
        s_sh = sqrtf(a * a + b2 * b2 + c * c + d * d);
    }
    __syncthreads();
    float s = s_sh;
    const float* vv = vec + (size_t)row * DD;
    float y0 = vv[t] + s * Ws[t] + bs[t];
    float y1 = vv[t + 256] + s * Ws[t + 256] + bs[t + 256];
    __shared__ float red[256];
    red[t] = y0 + y1;
    __syncthreads();
    for (int st = 128; st > 0; st >>= 1) { if (t < st) red[t] += red[t + st]; __syncthreads(); }
    float mean = red[0] * (1.0f / DD);
    __syncthreads();
    float d0 = y0 - mean, d1 = y1 - mean;
    red[t] = d0 * d0 + d1 * d1;
    __syncthreads();
    for (int st = 128; st > 0; st >>= 1) { if (t < st) red[t] += red[t + st]; __syncthreads(); }
    float rs = rsqrtf(red[0] * (1.0f / DD) + 1e-5f);
    hout[(size_t)row * DD + t]       = d0 * rs * g[t] + bp[t];
    hout[(size_t)row * DD + t + 256] = d1 * rs * g[t + 256] + bp[t + 256];
}

__global__ void k_vtranspose(const half_t* __restrict__ qkv, half_t* __restrict__ vT) {
    int tid = blockIdx.x * blockDim.x + threadIdx.x;
    if (tid >= BB * HHN * HDD * LLEN) return;
    int l = tid % LLEN;
    int r = tid / LLEN;
    int d = r % HDD;
    int r2 = r / HDD;
    int hh = r2 % HHN;
    int b = r2 / HHN;
    vT[tid] = qkv[((size_t)(b * LLEN + l)) * D3 + 2 * DD + hh * HDD + d];
}

__global__ void k_softmax_bias(const float* __restrict__ scores, const float* __restrict__ inter,
                               const float* __restrict__ Wip, half_t* __restrict__ attn) {
    int q = blockIdx.x % LLEN;
    int z = blockIdx.x / LLEN;
    int hh = z % HHN;
    int b = z / HHN;
    int k = threadIdx.x; // 128
    float sc = scores[((size_t)z * LLEN + q) * LLEN + k];
    const float* ip = inter + (((size_t)(b * LLEN + q)) * LLEN + k) * 4;
    const float* w = Wip + hh * 4;
    sc += ip[0] * w[0] + ip[1] * w[1] + ip[2] * w[2] + ip[3] * w[3];
    __shared__ float red[128];
    red[k] = sc;
    __syncthreads();
    for (int s = 64; s > 0; s >>= 1) { if (k < s) red[k] = fmaxf(red[k], red[k + s]); __syncthreads(); }
    float mx = red[0];
    __syncthreads();
    float e = expf(sc - mx);
    red[k] = e;
    __syncthreads();
    for (int s = 64; s > 0; s >>= 1) { if (k < s) red[k] += red[k + s]; __syncthreads(); }
    float sum = red[0];
    attn[((size_t)z * LLEN + q) * LLEN + k] = (half_t)(e / sum);
}

__global__ void k_pool(const float* __restrict__ h, float* __restrict__ pooled) {
    int tid = blockIdx.x * blockDim.x + threadIdx.x;
    if (tid >= BB * DD) return;
    int d = tid % DD;
    int b = tid / DD;
    float acc = 0.f;
    for (int l = 0; l < LLEN; ++l) acc += h[((size_t)(b * LLEN + l)) * DD + d];
    pooled[tid] = acc * (1.0f / LLEN);
}

__global__ void k_mlp_small(const float* __restrict__ in, const float* __restrict__ W,
                            const float* __restrict__ bias, float* __restrict__ out,
                            int M, int N, int K, int act) {
    int tid = blockIdx.x * blockDim.x + threadIdx.x;
    if (tid >= M * N) return;
    int n = tid % N;
    int m = tid / N;
    float acc = bias[n];
    const float* ir = in + (size_t)m * K;
    const float* wr = W + (size_t)n * K;
    for (int k = 0; k < K; ++k) acc += ir[k] * wr[k];
    if (act == 1) acc = gelu_exact(acc);
    else if (act == 2) acc = softplus_f(acc);
    out[tid] = acc;
}

__global__ void k_recon2(const half_t* __restrict__ r1, const float* __restrict__ W,
                         const float* __restrict__ bias, float* __restrict__ out) {
    int tid = blockIdx.x * blockDim.x + threadIdx.x;
    if (tid >= BB * LLEN * FEATN) return;
    int f = tid % FEATN;
    int m = tid / FEATN;
    float acc = bias[f];
    const half_t* ir = r1 + (size_t)m * DHALF;
    const float* wr = W + (size_t)f * DHALF;
    for (int k = 0; k < DHALF; ++k) acc += (float)ir[k] * wr[k];
    out[tid] = acc;
}

// ---------------------------------------------------------------------------
// host launcher
// ---------------------------------------------------------------------------
static inline int ceil_div(long a, long b) { return (int)((a + b - 1) / b); }

static inline void launch_gemm(const half_t* A, const half_t* W, const float* bias,
                               const float* res, float* Cf, half_t* Ch,
                               int M, int N, int K, int lda, int ldw, int ldc, int act,
                               hipStream_t stream) {
    int blocks = (M / 256) * (N / 64);
    k_gemm_tdm<<<blocks, 256, 0, stream>>>(A, W, bias, res, Cf, Ch,
                                           M, N, K, lda, ldw, ldc, act);
}

extern "C" void kernel_launch(void* const* d_in, const int* in_sizes, int n_in,
                              void* d_out, int out_size, void* d_ws, size_t ws_size,
                              hipStream_t stream) {
    const float* x       = (const float*)d_in[0];
    const float* W_in    = (const float*)d_in[1];
    const float* b_in    = (const float*)d_in[2];
    const float* pos     = (const float*)d_in[3];
    const float* pre_Wv  = (const float*)d_in[4];
    const float* pre_Ws  = (const float*)d_in[5];
    const float* pre_bs  = (const float*)d_in[6];
    const float* pre_g   = (const float*)d_in[7];
    const float* pre_b   = (const float*)d_in[8];
    const float* post_Wv = (const float*)d_in[9];
    const float* post_Ws = (const float*)d_in[10];
    const float* post_bs = (const float*)d_in[11];
    const float* post_g  = (const float*)d_in[12];
    const float* post_b  = (const float*)d_in[13];
    const float* ln1_g = (const float*)d_in[14];
    const float* ln1_b = (const float*)d_in[15];
    const float* Wqkv  = (const float*)d_in[16];
    const float* bqkv  = (const float*)d_in[17];
    const float* Wo    = (const float*)d_in[18];
    const float* bo    = (const float*)d_in[19];
    const float* Wip   = (const float*)d_in[20];
    const float* ln2_g = (const float*)d_in[21];
    const float* ln2_b = (const float*)d_in[22];
    const float* W1    = (const float*)d_in[23];
    const float* b1    = (const float*)d_in[24];
    const float* W2    = (const float*)d_in[25];
    const float* b2    = (const float*)d_in[26];
    const float* Wr1 = (const float*)d_in[27];
    const float* br1 = (const float*)d_in[28];
    const float* Wr2 = (const float*)d_in[29];
    const float* br2 = (const float*)d_in[30];
    const float* Wc1 = (const float*)d_in[31];
    const float* bc1 = (const float*)d_in[32];
    const float* Wc2 = (const float*)d_in[33];
    const float* bc2 = (const float*)d_in[34];
    const float* Wm1 = (const float*)d_in[35];
    const float* bm1 = (const float*)d_in[36];
    const float* Wm2 = (const float*)d_in[37];
    const float* bm2 = (const float*)d_in[38];

    float* out = (float*)d_out;
    float* out_recon = out;                                   // B*L*FEAT
    float* out_cls   = out + (size_t)BB * LLEN * FEATN;       // B*NCLS
    float* out_mass  = out_cls + (size_t)BB * NCLSN;          // B

    // workspace carve-up
    char* wsb = (char*)d_ws;
    size_t off = 0;
    auto alloc = [&](size_t bytes) -> void* {
        off = (off + 255) & ~(size_t)255;
        void* p = wsb + off;
        off += bytes;
        return p;
    };
    const size_t NTOK = (size_t)BB * LLEN;     // 8192
    float*  inter   = (float*)alloc((size_t)BB * LLEN * LLEN * 4 * 4);
    float*  h       = (float*)alloc(NTOK * DD * 4);
    float*  tmp     = (float*)alloc(NTOK * DD * 4);
    half_t* h16     = (half_t*)alloc(NTOK * DD * 2);
    half_t* hn16    = (half_t*)alloc(NTOK * DD * 2);
    half_t* qkv16   = (half_t*)alloc(NTOK * D3 * 2);
    half_t* vT16    = (half_t*)alloc(NTOK * DD * 2);
    float*  scores  = (float*)alloc((size_t)BB * HHN * LLEN * LLEN * 4);
    half_t* attn16  = (half_t*)alloc((size_t)BB * HHN * LLEN * LLEN * 2);
    half_t* o16     = (half_t*)alloc(NTOK * DD * 2);
    half_t* ff16    = (half_t*)alloc(NTOK * D4 * 2);
    half_t* r1_16   = (half_t*)alloc(NTOK * DHALF * 2);
    float*  pooled  = (float*)alloc((size_t)BB * DD * 4);
    float*  c1      = (float*)alloc((size_t)BB * DHALF * 4);
    float*  m1      = (float*)alloc((size_t)BB * DQRT * 4);
    half_t* wvpre16  = (half_t*)alloc((size_t)DD * DD * 2);
    half_t* wvpost16 = (half_t*)alloc((size_t)DD * DD * 2);
    half_t* wqkv16   = (half_t*)alloc((size_t)NLAYER * D3 * DD * 2);
    half_t* wo16     = (half_t*)alloc((size_t)NLAYER * DD * DD * 2);
    half_t* w1_16    = (half_t*)alloc((size_t)NLAYER * D4 * DD * 2);
    half_t* w2_16    = (half_t*)alloc((size_t)NLAYER * DD * D4 * 2);
    half_t* wr1_16   = (half_t*)alloc((size_t)DHALF * DD * 2);

    auto conv = [&](const float* s, half_t* dstp, long n) {
        k_f32_to_f16<<<ceil_div(n, 256), 256, 0, stream>>>(s, dstp, (int)n);
    };

    // ---- weight conversions (f32 -> f16) ----
    conv(pre_Wv,  wvpre16,  (long)DD * DD);
    conv(post_Wv, wvpost16, (long)DD * DD);
    conv(Wqkv,    wqkv16,   (long)NLAYER * D3 * DD);
    conv(Wo,      wo16,     (long)NLAYER * DD * DD);
    conv(W1,      w1_16,    (long)NLAYER * D4 * DD);
    conv(W2,      w2_16,    (long)NLAYER * DD * D4);
    conv(Wr1,     wr1_16,   (long)DHALF * DD);

    // ---- interaction matrix + embedding ----
    k_interaction<<<ceil_div((long)BB * LLEN * LLEN, 256), 256, 0, stream>>>(x, inter);
    k_embed<<<ceil_div((long)NTOK * DD, 256), 256, 0, stream>>>(x, W_in, b_in, pos, h);

    // ---- pre equi block: vec = h @ Wv^T ; LN(vec + s*Ws + bs) ----
    conv(h, h16, (long)NTOK * DD);
    launch_gemm(h16, wvpre16, nullptr, nullptr, tmp, nullptr,
                (int)NTOK, DD, DD, DD, DD, DD, 0, stream);
    k_equi_finish<<<(int)NTOK, 256, 0, stream>>>(tmp, h, pre_Ws, pre_bs, pre_g, pre_b, h);

    const float inv_scale = 0.125f; // 1/sqrt(64)

    // ---- transformer layers ----
    for (int l = 0; l < NLAYER; ++l) {
        k_ln<<<(int)NTOK, 256, 0, stream>>>(h, ln1_g + l * DD, ln1_b + l * DD, hn16);

        // qkv = LN(h) @ Wqkv^T + bqkv   (M=8192, N=1536, K=512) -> f16
        launch_gemm(hn16, wqkv16 + (size_t)l * D3 * DD, bqkv + l * D3, nullptr,
                    nullptr, qkv16, (int)NTOK, D3, DD, DD, DD, D3, 0, stream);
        k_vtranspose<<<ceil_div((long)NTOK * DD, 256), 256, 0, stream>>>(qkv16, vT16);

        // scores = (q @ k^T) / sqrt(hd)  per (b,h):  M=N=128, K=64
        {
            dim3 grid(2, 1, BB * HHN);
            k_gemm_batched<<<grid, 256, 0, stream>>>(
                qkv16 /*q cols 0..511*/, qkv16 + DD /*k cols 512..1023*/,
                scores, nullptr,
                LLEN, LLEN, HDD, D3, D3, LLEN,
                (long)LLEN * D3, (long)HDD, (long)LLEN * D3, (long)HDD,
                (long)HHN * LLEN * LLEN, (long)LLEN * LLEN,
                HHN, inv_scale);
        }

        // softmax with interaction bias -> attn (f16)
        k_softmax_bias<<<BB * HHN * LLEN, 128, 0, stream>>>(scores, inter,
                                                            Wip + l * HHN * 4, attn16);

        // o = attn @ v  per (b,h):  M=128, N=64, K=128 -> o16 (B,L,D) layout
        {
            dim3 grid(1, 1, BB * HHN);
            k_gemm_batched<<<grid, 256, 0, stream>>>(
                attn16, vT16, nullptr, o16,
                LLEN, HDD, LLEN, LLEN, LLEN, DD,
                (long)HHN * LLEN * LLEN, (long)LLEN * LLEN,
                (long)HHN * HDD * LLEN, (long)HDD * LLEN,
                (long)LLEN * DD, (long)HDD,
                HHN, 1.0f);
        }

        // h = h + o @ Wo^T + bo
        launch_gemm(o16, wo16 + (size_t)l * DD * DD, bo + l * DD, h,
                    h, nullptr, (int)NTOK, DD, DD, DD, DD, DD, 0, stream);

        // MLP: f = gelu(LN2(h) @ W1^T + b1) ; h = h + f @ W2^T + b2
        k_ln<<<(int)NTOK, 256, 0, stream>>>(h, ln2_g + l * DD, ln2_b + l * DD, hn16);
        launch_gemm(hn16, w1_16 + (size_t)l * D4 * DD, b1 + l * D4, nullptr,
                    nullptr, ff16, (int)NTOK, D4, DD, DD, DD, D4, 1, stream);
        launch_gemm(ff16, w2_16 + (size_t)l * DD * D4, b2 + l * DD, h,
                    h, nullptr, (int)NTOK, DD, D4, D4, D4, DD, 0, stream);
    }

    // ---- post equi block ----
    conv(h, h16, (long)NTOK * DD);
    launch_gemm(h16, wvpost16, nullptr, nullptr, tmp, nullptr,
                (int)NTOK, DD, DD, DD, DD, DD, 0, stream);
    k_equi_finish<<<(int)NTOK, 256, 0, stream>>>(tmp, h, post_Ws, post_bs, post_g, post_b, h);

    // ---- heads ----
    conv(h, h16, (long)NTOK * DD);
    // recon layer 1 (WMMA): gelu(h @ Wr1^T + br1) -> r1_16 (M=8192, N=256, K=512)
    launch_gemm(h16, wr1_16, br1, nullptr, nullptr, r1_16,
                (int)NTOK, DHALF, DD, DD, DD, DHALF, 1, stream);
    k_recon2<<<ceil_div((long)NTOK * FEATN, 256), 256, 0, stream>>>(r1_16, Wr2, br2, out_recon);

    // pooled mean over L
    k_pool<<<ceil_div((long)BB * DD, 256), 256, 0, stream>>>(h, pooled);

    // classifier head
    k_mlp_small<<<ceil_div((long)BB * DHALF, 256), 256, 0, stream>>>(pooled, Wc1, bc1, c1,
                                                                     BB, DHALF, DD, 1);
    k_mlp_small<<<ceil_div((long)BB * NCLSN, 256), 256, 0, stream>>>(c1, Wc2, bc2, out_cls,
                                                                     BB, NCLSN, DHALF, 0);
    // mass head
    k_mlp_small<<<ceil_div((long)BB * DQRT, 256), 256, 0, stream>>>(pooled, Wm1, bm1, m1,
                                                                    BB, DQRT, DD, 1);
    k_mlp_small<<<ceil_div((long)BB * 1, 256), 256, 0, stream>>>(m1, Wm2, bm2, out_mass,
                                                                 BB, 1, DQRT, 2);
}
